// deformable_cross_attention_46024869544003
// MI455X (gfx1250) — matrix operand order repaired and checked by reading, not compiled, not run on hardware
//
#include <hip/hip_runtime.h>
#include <hip/hip_bf16.h>

typedef _Float16 v16h __attribute__((ext_vector_type(16)));
typedef _Float16 v8h  __attribute__((ext_vector_type(8)));
typedef _Float16 v4h  __attribute__((ext_vector_type(4)));
typedef float    v8f  __attribute__((ext_vector_type(8)));

#define Qn   16384
#define Cc   128
#define Hh   96
#define Ww   352
#define HWp  (Hh * Ww)      // 33792 pixels per view
#define Nv   5
#define NSs  8
#define LDA  136            // padded LDS row pitch in halfs (16B-aligned rows)

// ---- WMMA fragment loaders from LDS tiles (ISA 7.12.2 layouts) -------------
// A (16x32 f16): lanes 0-15 rows M, VGPR0-3 K=0..7, VGPR4-7 K=16..23;
//                lanes 16-31 same rows, K=8..15 and K=24..31.
// B (32x16 f16): lanes 0-15 col N, K=0..7 and K=16..23; lanes 16-31 col N, K=8..15 and K=24..31.
__device__ __forceinline__ v16h make_frag(const _Float16* lo, const _Float16* hi) {
    v8h l = *(const v8h*)lo;
    v8h h = *(const v8h*)hi;
    v16h f;
#pragma unroll
    for (int i = 0; i < 8; ++i) { f[i] = l[i]; f[i + 8] = h[i]; }
    return f;
}
__device__ __forceinline__ v16h fragA(const _Float16* row, int kb, int g) {
    return make_frag(row + kb + 8 * g, row + kb + 16 + 8 * g);
}
__device__ __forceinline__ v16h fragB(const _Float16* row, int kb, int g) {
    return make_frag(row + kb + 8 * g, row + kb + 16 + 8 * g);
}
#define WMMA_F16(a, b, c) \
    __builtin_amdgcn_wmma_f32_16x16x32_f16(false, (a), false, (b), (short)0, (c), false, false)

// ---------------------------------------------------------------------------
// K1: per-query sampling offsets + attention softmax.
// delta_p = q @ w_dp + b_dp   (16 outs), attn = softmax(q @ w_a + b_a) (8)
// grid[n,q,ns,2] = 2*(ref[n,q] + delta_p[ns]) - 1
// ---------------------------------------------------------------------------
__global__ __launch_bounds__(256) void k_offsets(
    const float* __restrict__ query, const float* __restrict__ ref,
    const float* __restrict__ w_dp, const float* __restrict__ b_dp,
    const float* __restrict__ w_a,  const float* __restrict__ b_a,
    float* __restrict__ grid, float* __restrict__ attn)
{
    int q = blockIdx.x * blockDim.x + threadIdx.x;
    if (q >= Qn) return;
    float dp[16], aa[8];
#pragma unroll
    for (int j = 0; j < 16; ++j) dp[j] = b_dp[j];
#pragma unroll
    for (int j = 0; j < 8; ++j) aa[j] = b_a[j];
    const float* qr = query + (size_t)q * Cc;
    for (int c = 0; c < Cc; ++c) {
        float qv = qr[c];
        const float* wd = w_dp + c * 16;
        const float* wa = w_a + c * 8;
#pragma unroll
        for (int j = 0; j < 16; ++j) dp[j] = fmaf(qv, wd[j], dp[j]);
#pragma unroll
        for (int j = 0; j < 8; ++j) aa[j] = fmaf(qv, wa[j], aa[j]);
    }
    float mx = aa[0];
#pragma unroll
    for (int j = 1; j < 8; ++j) mx = fmaxf(mx, aa[j]);
    float s = 0.f;
#pragma unroll
    for (int j = 0; j < 8; ++j) { aa[j] = __expf(aa[j] - mx); s += aa[j]; }
    float inv = 1.f / s;
#pragma unroll
    for (int j = 0; j < 8; ++j) attn[(size_t)q * 8 + j] = aa[j] * inv;

    for (int n = 0; n < Nv; ++n) {
        float rx = ref[((size_t)n * Qn + q) * 2 + 0];
        float ry = ref[((size_t)n * Qn + q) * 2 + 1];
#pragma unroll
        for (int ns = 0; ns < NSs; ++ns) {
            size_t gi = (((size_t)n * Qn + q) * 8 + ns) * 2;
            grid[gi + 0] = 2.f * (rx + dp[ns * 2 + 0]) - 1.f;
            grid[gi + 1] = 2.f * (ry + dp[ns * 2 + 1]) - 1.f;
        }
    }
}

// ---------------------------------------------------------------------------
// K2: value projection v[bn,pix,d] = sum_c img[bn,c,pix]*w_v[c,d] + b_v[d]
// WMMA GEMM, output stored f16 pixel-major (43MB -> L2 resident for sampling).
// WG tile: 64 pixels x 128 d, 8 waves (4 pixel sub-tiles x 2 d halves).
// ---------------------------------------------------------------------------
__global__ __launch_bounds__(256) void k_vproj(
    const float* __restrict__ img, const float* __restrict__ w_v,
    const float* __restrict__ b_v, _Float16* __restrict__ vout)
{
    __shared__ _Float16 sA[64 * LDA];    // [pix][c]
    __shared__ _Float16 sB[128 * LDA];   // [d][c]  (w_v transposed)
    const int tid = threadIdx.x;
    const int bn = blockIdx.x / (HWp / 64);
    const int pixbase = (blockIdx.x % (HWp / 64)) * 64;
    const float* imgv = img + (size_t)bn * Cc * HWp;

    for (int i = tid; i < 128 * 128; i += 256) {
        int c = i >> 7, d = i & 127;
        sB[d * LDA + c] = (_Float16)w_v[i];
    }
    for (int i = tid; i < 64 * 128; i += 256) {
        int c = i >> 6, p = i & 63;
        sA[p * LDA + c] = (_Float16)imgv[(size_t)c * HWp + pixbase + p];
    }
    __syncthreads();

    const int lane = tid & 31, wid = tid >> 5;
    const int m = lane & 15, g = lane >> 4, nn = lane & 15;
    const int pixw = (wid & 3) * 16;
    const int dt0 = (wid >> 2) * 4;

    v8f acc[4] = {};
#pragma unroll
    for (int kb = 0; kb < 128; kb += 32) {
        v16h a = fragA(&sA[(pixw + m) * LDA], kb, g);
#pragma unroll
        for (int t = 0; t < 4; ++t) {
            v16h b = fragB(&sB[((dt0 + t) * 16 + nn) * LDA], kb, g);
            acc[t] = WMMA_F16(a, b, acc[t]);
        }
    }
#pragma unroll
    for (int t = 0; t < 4; ++t) {
        int d = (dt0 + t) * 16 + nn;
        float bv = b_v[d];
#pragma unroll
        for (int r = 0; r < 8; ++r) {
            int pix = pixbase + pixw + r + 8 * g;
            vout[((size_t)bn * HWp + pix) * Cc + d] = (_Float16)(acc[t][r] + bv);
        }
    }
}

// ---------------------------------------------------------------------------
// K3: fused bilinear sampling + attn weighting + mask accumulation.
// One wave per query; lanes own 4 channels each. All gathers hit the
// L2-resident f16 value tensor; each corner = 256B contiguous per wave.
// t[q,c] = sum_n m[n,q] * sum_ns attn[q,ns] * bilinear(v[n], grid[n,q,ns])[c]
// ---------------------------------------------------------------------------
__device__ __forceinline__ void corner_acc(const _Float16* vb, float xf, float yf,
                                           float w, float acc[4], int lane)
{
    bool valid = (xf >= 0.f) & (xf <= (float)(Ww - 1)) &
                 (yf >= 0.f) & (yf <= (float)(Hh - 1));
    float wv = valid ? w : 0.f;
    int xi = (int)xf; xi = xi < 0 ? 0 : (xi > Ww - 1 ? Ww - 1 : xi);
    int yi = (int)yf; yi = yi < 0 ? 0 : (yi > Hh - 1 ? Hh - 1 : yi);
    v4h d = *(const v4h*)(vb + ((size_t)yi * Ww + xi) * Cc + lane * 4);
#pragma unroll
    for (int j = 0; j < 4; ++j) acc[j] = fmaf(wv, (float)d[j], acc[j]);
}

__global__ __launch_bounds__(256) void k_sample(
    const _Float16* __restrict__ v, const float* __restrict__ grid,
    const float* __restrict__ attn, const int* __restrict__ mask,
    _Float16* __restrict__ tout, float* __restrict__ smout)
{
    const int lane = threadIdx.x & 31;
    const int q = blockIdx.x * 8 + (threadIdx.x >> 5);
    float acc[4] = {0.f, 0.f, 0.f, 0.f};
    float aw[8];
#pragma unroll
    for (int s = 0; s < 8; ++s) aw[s] = attn[(size_t)q * 8 + s];
    float msum = 0.f;
    for (int n = 0; n < Nv; ++n) {
        float fm = (float)mask[(size_t)n * Qn + q];
        if (fm == 0.f) continue;          // wave-uniform branch (q uniform/wave)
        msum += fm;
        const _Float16* vb = v + (size_t)n * HWp * Cc;
#pragma unroll
        for (int s = 0; s < NSs; ++s) {
            size_t gi = (((size_t)n * Qn + q) * 8 + s) * 2;
            float fx = (grid[gi + 0] + 1.f) * 0.5f * (float)(Ww - 1);
            float fy = (grid[gi + 1] + 1.f) * 0.5f * (float)(Hh - 1);
            float x0 = floorf(fx), y0 = floorf(fy);
            float wx1 = fx - x0, wy1 = fy - y0;
            float wx0 = 1.f - wx1, wy0 = 1.f - wy1;
            float as = aw[s];
            corner_acc(vb, x0,       y0,       as * wx0 * wy0, acc, lane);
            corner_acc(vb, x0 + 1.f, y0,       as * wx1 * wy0, acc, lane);
            corner_acc(vb, x0,       y0 + 1.f, as * wx0 * wy1, acc, lane);
            corner_acc(vb, x0 + 1.f, y0 + 1.f, as * wx1 * wy1, acc, lane);
        }
    }
#pragma unroll
    for (int j = 0; j < 4; ++j)
        tout[(size_t)q * Cc + lane * 4 + j] = (_Float16)acc[j];
    if (lane == 0) smout[q] = msum;
}

// ---------------------------------------------------------------------------
// K4: fused double GEMM.
// u = (t @ w_w + sm*b_w) / (sm + 1e-6);  out = u @ w_o + b_o
// WG tile: 64 queries x 128 outs; u written in place over the t tile in LDS.
// ---------------------------------------------------------------------------
__global__ __launch_bounds__(256) void k_output(
    const _Float16* __restrict__ t, const float* __restrict__ sm,
    const float* __restrict__ w_w, const float* __restrict__ b_w,
    const float* __restrict__ w_o, const float* __restrict__ b_o,
    float* __restrict__ out)
{
    __shared__ _Float16 sT[64 * LDA];    // [q][c] -> later [q][d] (= u)
    __shared__ _Float16 sW[128 * LDA];   // [out][in] transposed weights
    const int tid = threadIdx.x;
    const int qbase = blockIdx.x * 64;

    for (int i = tid; i < 64 * 128; i += 256) {
        int p = i >> 7, c = i & 127;
        sT[p * LDA + c] = t[(size_t)(qbase + p) * Cc + c];
    }
    for (int i = tid; i < 128 * 128; i += 256) {
        int c = i >> 7, d = i & 127;
        sW[d * LDA + c] = (_Float16)w_w[i];
    }
    __syncthreads();

    const int lane = tid & 31, wid = tid >> 5;
    const int m = lane & 15, g = lane >> 4, nn = lane & 15;
    const int qw = (wid & 3) * 16;
    const int dt0 = (wid >> 2) * 4;

    // pass 1: t @ w_w
    v8f acc[4] = {};
#pragma unroll
    for (int kb = 0; kb < 128; kb += 32) {
        v16h a = fragA(&sT[(qw + m) * LDA], kb, g);
#pragma unroll
        for (int td = 0; td < 4; ++td) {
            v16h b = fragB(&sW[((dt0 + td) * 16 + nn) * LDA], kb, g);
            acc[td] = WMMA_F16(a, b, acc[td]);
        }
    }
    __syncthreads();   // all reads of sT/sW(w_w) complete

    // normalize + bias, write u over sT
#pragma unroll
    for (int td = 0; td < 4; ++td) {
        int d = (dt0 + td) * 16 + nn;
        float bw = b_w[d];
#pragma unroll
        for (int r = 0; r < 8; ++r) {
            int qr = qw + r + 8 * g;
            float smv = sm[qbase + qr];
            sT[qr * LDA + d] = (_Float16)((acc[td][r] + smv * bw) / (smv + 1e-6f));
        }
    }
    __syncthreads();

    // restage w_o transposed
    for (int i = tid; i < 128 * 128; i += 256) {
        int c = i >> 7, d = i & 127;
        sW[d * LDA + c] = (_Float16)w_o[i];
    }
    __syncthreads();

    // pass 2: u @ w_o
    v8f acc2[4] = {};
#pragma unroll
    for (int kb = 0; kb < 128; kb += 32) {
        v16h a = fragA(&sT[(qw + m) * LDA], kb, g);
#pragma unroll
        for (int td = 0; td < 4; ++td) {
            v16h b = fragB(&sW[((dt0 + td) * 16 + nn) * LDA], kb, g);
            acc2[td] = WMMA_F16(a, b, acc2[td]);
        }
    }
#pragma unroll
    for (int td = 0; td < 4; ++td) {
        int d = (dt0 + td) * 16 + nn;
        float bo = b_o[d];
#pragma unroll
        for (int r = 0; r < 8; ++r) {
            int qr = qw + r + 8 * g;
            out[(size_t)(qbase + qr) * Cc + d] = acc2[td][r] + bo;
        }
    }
}

// ---------------------------------------------------------------------------
extern "C" void kernel_launch(void* const* d_in, const int* in_sizes, int n_in,
                              void* d_out, int out_size, void* d_ws, size_t ws_size,
                              hipStream_t stream)
{
    const float* query = (const float*)d_in[0];
    const float* ref   = (const float*)d_in[1];
    const float* img   = (const float*)d_in[2];
    const int*   mask  = (const int*)d_in[3];
    const float* w_dp  = (const float*)d_in[4];
    const float* b_dp  = (const float*)d_in[5];
    const float* w_a   = (const float*)d_in[6];
    const float* b_a   = (const float*)d_in[7];
    const float* w_w   = (const float*)d_in[8];
    const float* b_w   = (const float*)d_in[9];
    const float* w_v   = (const float*)d_in[10];
    const float* b_v   = (const float*)d_in[11];
    const float* w_o   = (const float*)d_in[12];
    const float* b_o   = (const float*)d_in[13];
    float* out = (float*)d_out;

    // workspace layout (all 256B aligned)
    char* ws = (char*)d_ws;
    constexpr size_t vbytes    = (size_t)Nv * HWp * Cc * 2;        // 43,253,760
    constexpr size_t gridbytes = (size_t)Nv * Qn * NSs * 2 * 4;    //  5,242,880
    constexpr size_t attnbytes = (size_t)Qn * NSs * 4;             //    524,288
    constexpr size_t tbytes    = (size_t)Qn * Cc * 2;              //  4,194,304
    _Float16* vws   = (_Float16*)(ws);
    float*    gridw = (float*)(ws + vbytes);
    float*    attnw = (float*)(ws + vbytes + gridbytes);
    _Float16* tws   = (_Float16*)(ws + vbytes + gridbytes + attnbytes);
    float*    smws  = (float*)(ws + vbytes + gridbytes + attnbytes + tbytes);

    k_offsets<<<Qn / 256, 256, 0, stream>>>(query, ref, w_dp, b_dp, w_a, b_a,
                                            gridw, attnw);
    k_vproj<<<Nv * (HWp / 64), 256, 0, stream>>>(img, w_v, b_v, vws);
    k_sample<<<Qn / 8, 256, 0, stream>>>(vws, gridw, attnw, mask, tws, smws);
    k_output<<<Qn / 64, 256, 0, stream>>>(tws, smws, w_w, b_w, w_o, b_o, out);
}
